// DualCollaborativeLevelEncoder_9835475108510
// MI455X (gfx1250) — matrix operand order repaired and checked by reading, not compiled, not run on hardware
//
#include <hip/hip_runtime.h>
#include <hip/hip_bf16.h>
#include <cstdint>
#include <cstddef>

typedef __attribute__((ext_vector_type(16))) _Float16 v16h;
typedef __attribute__((ext_vector_type(8)))  float    v8f;

namespace cfg {
constexpr int B = 32, NR = 100, NG = 49, NA = 149;
constexpr int DIN = 2048, DM = 512, H = 8, DK = 64, DFF = 2048, NL = 3, GRID = 7;
constexpr float NEGF = -1e9f;
}

// ---------------------------------------------------------------------------
// WMMA GEMM:  C[M,N] = act(A[M,K] * W[K,N] + bias)   A,W fp32 -> f16 tiles in LDS
// Block tile 128x128x32, 256 threads = 8 waves (2x4), wave tile 64x32.
// - A tile M-major [m][k]; W tile TRANSPOSED [n][k]: both fragment gathers are
//   two contiguous 8-half runs -> ds_load_b128 pairs.
// - Double-buffered LDS: one barrier per K-step; staging issues ALL global
//   loads first (batched, one loadcnt wait), then converts+stores, so the
//   next tile's fetch overlaps the WMMA burst.
// - A rows are CLAMPED to M-1 instead of zero-guarded: an A row only feeds its
//   own output row, and OOB output rows are dropped in the epilogue, so the
//   staging path is branch-free for every block.
// ---------------------------------------------------------------------------
#define GBM 128
#define GBN 128
#define GBK 32
#define LPAD 16   // stride 48 halves = 96B rows: keeps b128 fragment loads 16B-aligned

template <bool RELU>
__global__ __launch_bounds__(256) void gemm_bias_kernel(
    const float* __restrict__ A, const float* __restrict__ W,
    const float* __restrict__ bias, float* __restrict__ C,
    int M, int N, int K) {
  __shared__ _Float16 As[2][GBM][GBK + LPAD];
  __shared__ _Float16 Bs[2][GBN][GBK + LPAD];   // transposed: [col][k]

  const int tid  = threadIdx.x;
  const int lane = tid & 31;
  const int wid  = tid >> 5;
  const int wm   = (wid >> 2) * 64;   // 0 or 64
  const int wn   = (wid & 3) * 32;    // 0,32,64,96
  const int bm   = blockIdx.y * GBM;
  const int bn   = blockIdx.x * GBN;

  v8f acc[4][2];
  #pragma unroll
  for (int i = 0; i < 4; ++i)
    #pragma unroll
    for (int j = 0; j < 2; ++j)
      #pragma unroll
      for (int e = 0; e < 8; ++e) acc[i][j][e] = 0.0f;

  auto stage = [&](int buf, int k0) {
    // ---- phase 1: issue ALL global loads (batched; no dependent use yet) ----
    float4 va[4], w0[2], w1[2];
    #pragma unroll
    for (int it = 0; it < 4; ++it) {
      int idx = tid + 256 * it;           // < 1024 float4 slots
      int r = idx >> 3;                   // 0..127
      int c = (idx & 7) << 2;             // 0..28
      int grow = bm + r;
      if (grow >= M) grow = M - 1;        // clamp (see header note)
      va[it] = *reinterpret_cast<const float4*>(A + (size_t)grow * K + k0 + c);
    }
    #pragma unroll
    for (int it = 0; it < 2; ++it) {
      int idx = tid + 256 * it;           // < 512
      int kp  = idx >> 5;                 // 0..15 (k pair)
      int c4  = (idx & 31) << 2;          // 0..124
      const float* wp = W + (size_t)(k0 + 2 * kp) * N + bn + c4;
      w0[it] = *reinterpret_cast<const float4*>(wp);
      w1[it] = *reinterpret_cast<const float4*>(wp + N);
    }
    // ---- phase 2: convert fp32 -> f16 and store to LDS ----
    #pragma unroll
    for (int it = 0; it < 4; ++it) {
      int idx = tid + 256 * it;
      int r = idx >> 3;
      int c = (idx & 7) << 2;
      As[buf][r][c + 0] = (_Float16)va[it].x; As[buf][r][c + 1] = (_Float16)va[it].y;
      As[buf][r][c + 2] = (_Float16)va[it].z; As[buf][r][c + 3] = (_Float16)va[it].w;
    }
    #pragma unroll
    for (int it = 0; it < 2; ++it) {
      int idx = tid + 256 * it;
      int kp  = idx >> 5;
      int c4  = (idx & 31) << 2;
      int kk  = kp << 1;
      Bs[buf][c4 + 0][kk] = (_Float16)w0[it].x; Bs[buf][c4 + 0][kk + 1] = (_Float16)w1[it].x;
      Bs[buf][c4 + 1][kk] = (_Float16)w0[it].y; Bs[buf][c4 + 1][kk + 1] = (_Float16)w1[it].y;
      Bs[buf][c4 + 2][kk] = (_Float16)w0[it].z; Bs[buf][c4 + 2][kk + 1] = (_Float16)w1[it].z;
      Bs[buf][c4 + 3][kk] = (_Float16)w0[it].w; Bs[buf][c4 + 3][kk + 1] = (_Float16)w1[it].w;
    }
  };

  stage(0, 0);
  int p = 0;
  for (int k0 = 0; k0 < K; k0 += GBK) {
    __syncthreads();                       // staged buffer p ready; buffer p^1 free
    if (k0 + GBK < K) stage(p ^ 1, k0 + GBK);
    if (k0 + 2 * GBK < K)                  // reach ahead of the double buffer (global_prefetch_b8)
      __builtin_prefetch((const void*)(W + (size_t)(k0 + 2 * GBK + (tid >> 5)) * N + bn + ((tid & 31) << 2)), 0, 1);

    // --- fragments per documented CDNA5 16-bit layouts ---
    v16h afrag[4], bfrag[2];
    const int khalf = (lane >> 4) << 3;    // A: lanes 16..31 -> K+8
    #pragma unroll
    for (int mi = 0; mi < 4; ++mi) {
      int mrow = wm + 16 * mi + (lane & 15);
      #pragma unroll
      for (int v = 0; v < 8; ++v) {
        int kb = ((v & 4) << 2) + ((v & 3) << 1) + khalf; // {0..7|16..23} + lanegrp*8
        afrag[mi][2 * v]     = As[p][mrow][kb];
        afrag[mi][2 * v + 1] = As[p][mrow][kb + 1];
      }
    }
    const int kboff = (lane >> 4) << 3;    // B: lanes 16..31 -> K+8
    #pragma unroll
    for (int ni = 0; ni < 2; ++ni) {
      int ncol = wn + 16 * ni + (lane & 15);
      #pragma unroll
      for (int v = 0; v < 8; ++v) {
        int kb = ((v & 4) << 2) + ((v & 3) << 1) + kboff; // B: {0..7|16..23} + lanegrp*8
        bfrag[ni][2 * v]     = Bs[p][ncol][kb];
        bfrag[ni][2 * v + 1] = Bs[p][ncol][kb + 1];
      }
    }
    #pragma unroll
    for (int mi = 0; mi < 4; ++mi)
      #pragma unroll
      for (int ni = 0; ni < 2; ++ni)
        acc[mi][ni] = __builtin_amdgcn_wmma_f32_16x16x32_f16(
            false, afrag[mi], false, bfrag[ni], (short)0, acc[mi][ni], false, false);
    p ^= 1;
  }

  // --- epilogue: bias (+ReLU), C layout: VGPR r -> M = r (+8 for lanes>=16) ---
  #pragma unroll
  for (int mi = 0; mi < 4; ++mi) {
    #pragma unroll
    for (int ni = 0; ni < 2; ++ni) {
      int col = bn + wn + 16 * ni + (lane & 15);
      float bv = bias ? bias[col] : 0.0f;
      #pragma unroll
      for (int r = 0; r < 8; ++r) {
        int row = bm + wm + 16 * mi + r + ((lane >> 4) << 3);
        if (row < M) {
          float val = acc[mi][ni][r] + bv;
          if (RELU) val = fmaxf(val, 0.0f);
          C[(size_t)row * N + col] = val;
        }
      }
    }
  }
}

// ---------------------------------------------------------------------------
// Attention: one wave per (b,h,q) row. scores = qk/8 + geo_log, mask -> -1e9,
// softmax over nk (<=149), then out = a @ V.   Wave32 shuffle reductions.
// ---------------------------------------------------------------------------
__global__ __launch_bounds__(256) void attn_kernel(
    const float* __restrict__ Q, const float* __restrict__ Km,
    const float* __restrict__ Vm, const float* __restrict__ geo,
    const unsigned char* __restrict__ mask, float* __restrict__ Out,
    int nq, int nk, int gq0, int gk0, int mqs, int mbs) {
  using namespace cfg;
  __shared__ float qsh[8][64];
  __shared__ float ash[8][160];
  const int wid = threadIdx.x >> 5, lane = threadIdx.x & 31;
  const int total = B * H * nq;
  int row = blockIdx.x * 8 + wid;
  const bool active = (row < total);
  if (!active) row = total - 1;
  const int qi = row % nq;
  const int bh = row / nq;
  const int h = bh % H, b = bh / H;

  const float* qp = Q + ((size_t)(b * nq + qi) * H + h) * DK;
  qsh[wid][lane]      = qp[lane];
  qsh[wid][lane + 32] = qp[lane + 32];
  __syncthreads();

  const float* kbase = Km + (size_t)b * nk * (H * DK);
  const float* gbase = geo + (((size_t)(b * H + h)) * NA + gq0 + qi) * NA + gk0;
  const unsigned char* mbase = mask + (size_t)b * mbs + (size_t)qi * mqs;

  const int kcount = (nk + 31) >> 5;   // <= 5
  float s[5];
  float smax = -3.4e38f;
  for (int i = 0; i < kcount; ++i) {
    int k = lane + (i << 5);
    float val = NEGF;
    if (k < nk) {
      const float* kp = kbase + ((size_t)k * H + h) * DK;
      float dot = 0.f;
      #pragma unroll
      for (int d = 0; d < 64; d += 4) {
        float4 kv = *reinterpret_cast<const float4*>(kp + d);
        dot += kv.x * qsh[wid][d] + kv.y * qsh[wid][d + 1] +
               kv.z * qsh[wid][d + 2] + kv.w * qsh[wid][d + 3];
      }
      val = dot * 0.125f + gbase[k];
      if (mbase[k]) val = NEGF;
    }
    s[i] = val;
    smax = fmaxf(smax, val);
  }
  for (int off = 16; off; off >>= 1) smax = fmaxf(smax, __shfl_xor(smax, off, 32));
  float sum = 0.f;
  for (int i = 0; i < kcount; ++i) { float e = __expf(s[i] - smax); s[i] = e; sum += e; }
  for (int off = 16; off; off >>= 1) sum += __shfl_xor(sum, off, 32);
  float inv = 1.0f / sum;
  for (int i = 0; i < kcount; ++i) {
    int k = lane + (i << 5);
    if (k < 160) ash[wid][k] = s[i] * inv;
  }
  __syncthreads();

  float a0 = 0.f, a1 = 0.f;
  const float* vbase = Vm + (size_t)b * nk * (H * DK) + (size_t)h * DK;
  for (int k = 0; k < nk; ++k) {
    float a = ash[wid][k];
    const float* vp = vbase + (size_t)k * (H * DK);
    a0 += a * vp[lane];
    a1 += a * vp[lane + 32];
  }
  if (active) {
    float* op = Out + ((size_t)(b * nq + qi) * H + h) * DK;
    op[lane] = a0; op[lane + 32] = a1;
  }
}

// ---------------------------------------------------------------------------
// LayerNorm(x [+ res]) over 512, one wave per row.
// ---------------------------------------------------------------------------
__global__ __launch_bounds__(256) void add_ln_kernel(
    const float* __restrict__ X, const float* __restrict__ Rres,
    const float* __restrict__ g, const float* __restrict__ bta,
    float* __restrict__ Out, int rows) {
  const int wid = threadIdx.x >> 5, lane = threadIdx.x & 31;
  const int row = blockIdx.x * 8 + wid;
  if (row >= rows) return;
  const float* xp = X + (size_t)row * 512;
  const float* rp = Rres ? Rres + (size_t)row * 512 : nullptr;
  float v[16];
  float sum = 0.f;
  #pragma unroll
  for (int i = 0; i < 16; ++i) {
    int d = lane + 32 * i;
    float t = xp[d];
    if (rp) t += rp[d];
    v[i] = t; sum += t;
  }
  for (int off = 16; off; off >>= 1) sum += __shfl_xor(sum, off, 32);
  float mean = sum * (1.0f / 512.0f);
  float var = 0.f;
  #pragma unroll
  for (int i = 0; i < 16; ++i) { float d = v[i] - mean; var += d * d; }
  for (int off = 16; off; off >>= 1) var += __shfl_xor(var, off, 32);
  float rstd = rsqrtf(var * (1.0f / 512.0f) + 1e-5f);
  #pragma unroll
  for (int i = 0; i < 16; ++i) {
    int d = lane + 32 * i;
    Out[(size_t)row * 512 + d] = (v[i] - mean) * rstd * g[d] + bta[d];
  }
}

// ---------------------------------------------------------------------------
// Elementwise helpers
// ---------------------------------------------------------------------------
__global__ void add_pos_kernel(const float* __restrict__ X, const float* __restrict__ pos,
                               float* __restrict__ Out, int rows, int s) {
  int i = blockIdx.x * 256 + threadIdx.x;
  if (i >= rows * 512) return;
  int d = i & 511, row = i >> 9, si = row % s;
  Out[i] = X[i] + pos[(si << 9) + d];
}

__global__ void concat_pos_kernel(const float* __restrict__ Xr, const float* __restrict__ Xg,
                                  const float* __restrict__ pos, float* __restrict__ Out) {
  using namespace cfg;
  int i = blockIdx.x * 256 + threadIdx.x;
  if (i >= B * NA * 512) return;
  int d = i & 511, row = i >> 9, si = row % NA, b = row / NA;
  float v = (si < NR) ? Xr[((size_t)b * NR + si) * 512 + d]
                      : Xg[((size_t)b * NG + (si - NR)) * 512 + d];
  Out[i] = v + pos[(si << 9) + d];
}

__global__ void pos_embed_kernel(float* __restrict__ out, int s) {
  int i = blockIdx.x * 256 + threadIdx.x;
  if (i >= s * 512) return;
  int si = i >> 9, d = i & 511;
  float pseq = (float)(si + 1) / ((float)s + 1e-6f) * 6.2831853071795864f;
  int j2 = (d >> 1) << 1;
  float dim_t = __powf(10000.0f, (float)j2 / 512.0f);
  float p = pseq / dim_t;
  out[i] = (d & 1) ? __cosf(p) : __sinf(p);
}

__device__ inline void load_box(const float* rb, const float* gb, int b, int idx, float* o) {
  using namespace cfg;
  const float* p = (idx < NR) ? rb + ((size_t)b * NR + idx) * 4
                              : gb + ((size_t)b * NG + (idx - NR)) * 4;
  o[0] = p[0]; o[1] = p[1]; o[2] = p[2]; o[3] = p[3];
}

// geo_log (B,H,149,149) = log(clip(relu(emb @ gsw.T + gsb), 1e-6))
__global__ void geo_kernel(const float* __restrict__ rb, const float* __restrict__ gb,
                           const float* __restrict__ gsw, const float* __restrict__ gsb,
                           float* __restrict__ geo) {
  using namespace cfg;
  int i = blockIdx.x * 256 + threadIdx.x;
  if (i >= B * NA * NA) return;
  int k = i % NA, t = i / NA, q = t % NA, b = t / NA;
  float bq[4], bk[4];
  load_box(rb, gb, b, q, bq);
  load_box(rb, gb, b, k, bk);
  float cxq = (bq[0] + bq[2]) * 0.5f, cyq = (bq[1] + bq[3]) * 0.5f;
  float wq = bq[2] - bq[0] + 1.0f, hq = bq[3] - bq[1] + 1.0f;
  float cxk = (bk[0] + bk[2]) * 0.5f, cyk = (bk[1] + bk[3]) * 0.5f;
  float wk = bk[2] - bk[0] + 1.0f, hk = bk[3] - bk[1] + 1.0f;
  float pos[4];
  pos[0] = __logf(fmaxf(fabsf((cxq - cxk) / wq), 1e-3f));
  pos[1] = __logf(fmaxf(fabsf((cyq - cyk) / hq), 1e-3f));
  pos[2] = __logf(wq / wk);
  pos[3] = __logf(hq / hk);
  float se[32], ce[32];
  #pragma unroll
  for (int m = 0; m < 4; ++m)
    #pragma unroll
    for (int j = 0; j < 8; ++j) {
      float mul = 100.0f * pos[m] / __powf(1000.0f, (float)j * 0.125f);
      se[m * 8 + j] = __sinf(mul);
      ce[m * 8 + j] = __cosf(mul);
    }
  #pragma unroll
  for (int h = 0; h < 8; ++h) {
    const float* w = gsw + h * 64;
    float acc = gsb[h];
    #pragma unroll
    for (int d = 0; d < 32; ++d) acc += se[d] * w[d] + ce[d] * w[32 + d];
    geo[(((size_t)b * 8 + h) * NA + q) * NA + k] = __logf(fmaxf(fmaxf(acc, 0.0f), 1e-6f));
  }
}

__global__ __launch_bounds__(256) void pad_kernel(const float* __restrict__ feat,
                                                  unsigned char* __restrict__ pad, int rows) {
  const int wid = threadIdx.x >> 5, lane = threadIdx.x & 31;
  const int row = blockIdx.x * 8 + wid;
  if (row >= rows) return;
  const float4* p = reinterpret_cast<const float4*>(feat + (size_t)row * 2048);
  float s = 0.f;
  #pragma unroll
  for (int i = 0; i < 16; ++i) {
    float4 v = p[lane + 32 * i];
    s += v.x + v.y + v.z + v.w;
  }
  for (int off = 16; off; off >>= 1) s += __shfl_xor(s, off, 32);
  if (lane == 0) pad[row] = (s == 0.0f) ? 1 : 0;
}

__device__ inline unsigned char not_overlap(const float* bx, int kg) {
  float c = (float)(kg % 7), r = (float)(kg / 7);
  const float inv7 = 1.0f / 7.0f;
  float gx1 = c * inv7, gy1 = r * inv7, gx2 = (c + 1.f) * inv7, gy2 = (r + 1.f) * inv7;
  bool ov = (bx[0] < gx2) && (bx[2] > gx1) && (bx[1] < gy2) && (bx[3] > gy1);
  return ov ? 0 : 1;
}

__global__ void mask_r2all_kernel(const float* __restrict__ rb,
                                  const unsigned char* __restrict__ pad_r,
                                  unsigned char* __restrict__ m) {
  using namespace cfg;
  int i = blockIdx.x * 256 + threadIdx.x;
  if (i >= B * NR * NA) return;
  int k = i % NA, t = i / NA, q = t % NR, b = t / NR;
  unsigned char o;
  if (k < NR) o = pad_r[b * NR + k];
  else {
    const float* bx = rb + ((size_t)b * NR + q) * 4;
    o = not_overlap(bx, k - NR);
  }
  m[i] = o;
}

__global__ void mask_g2all_kernel(const float* __restrict__ rb,
                                  const unsigned char* __restrict__ pad_g,
                                  unsigned char* __restrict__ m) {
  using namespace cfg;
  int i = blockIdx.x * 256 + threadIdx.x;
  if (i >= B * NG * NA) return;
  int k = i % NA, t = i / NA, qg = t % NG, b = t / NG;
  unsigned char o;
  if (k < NR) {
    const float* bx = rb + ((size_t)b * NR + k) * 4;
    o = not_overlap(bx, qg);
  } else {
    o = pad_g[b * NG + (k - NR)];
  }
  m[i] = o;
}

__global__ void finalize_kernel(const float* __restrict__ xr, const float* __restrict__ xg,
                                const unsigned char* __restrict__ pr,
                                const unsigned char* __restrict__ pg,
                                const float* __restrict__ pos_r, float* __restrict__ out) {
  using namespace cfg;
  constexpr int O1 = B * NA * DM;
  constexpr int O2 = B * NA;
  constexpr int O3 = B * NR * DM;
  int i = blockIdx.x * 256 + threadIdx.x;
  if (i < O1) {
    int d = i & 511, row = i >> 9, si = row % NA, b = row / NA;
    out[i] = (si < NR) ? xr[((size_t)b * NR + si) * 512 + d]
                       : xg[((size_t)b * NG + (si - NR)) * 512 + d];
  } else if (i < O1 + O2) {
    int j = i - O1, si = j % NA, b = j / NA;
    out[i] = (si < NR) ? (pr[b * NR + si] ? 1.0f : 0.0f)
                       : (pg[b * NG + (si - NR)] ? 1.0f : 0.0f);
  } else if (i < O1 + O2 + O3) {
    int j = i - O1 - O2, d = j & 511, row = j >> 9, si = row % NR;
    out[i] = pos_r[(si << 9) + d];
  }
}

// ---------------------------------------------------------------------------
// Host orchestration
// ---------------------------------------------------------------------------
static inline int ceil_div(int a, int b) { return (a + b - 1) / b; }

struct StackP {
  const float *b1, *b2, *bk, *bo, *bq, *bv, *ln1b, *ln1g, *ln2b, *ln2g,
              *w1, *w2, *wk, *wo, *wq, *wv;
};

// stack leaves arrive sorted: b1,b2,bk,bo,bq,bv,ln1_b,ln1_g,ln2_b,ln2_g,w1,w2,wk,wo,wq,wv
static StackP get_stack(void* const* d_in, int base, int L) {
  using namespace cfg;
  auto f = [&](int i) { return (const float*)d_in[base + i]; };
  StackP s;
  s.b1 = f(0) + (size_t)L * DFF;   s.b2 = f(1) + (size_t)L * DM;
  s.bk = f(2) + (size_t)L * DM;    s.bo = f(3) + (size_t)L * DM;
  s.bq = f(4) + (size_t)L * DM;    s.bv = f(5) + (size_t)L * DM;
  s.ln1b = f(6) + (size_t)L * DM;  s.ln1g = f(7) + (size_t)L * DM;
  s.ln2b = f(8) + (size_t)L * DM;  s.ln2g = f(9) + (size_t)L * DM;
  s.w1 = f(10) + (size_t)L * DM * DFF;
  s.w2 = f(11) + (size_t)L * DFF * DM;
  s.wk = f(12) + (size_t)L * DM * DM;
  s.wo = f(13) + (size_t)L * DM * DM;
  s.wq = f(14) + (size_t)L * DM * DM;
  s.wv = f(15) + (size_t)L * DM * DM;
  return s;
}

static void enc_layer(hipStream_t stream, const float* qin, const float* kv, int nq, int nkv,
                      const float* geo, int gq0, int gk0,
                      const unsigned char* mask, int mqs, int mbs,
                      const StackP& S, float* out,
                      float* t_q, float* t_k, float* t_v, float* t_att,
                      float* t_o, float* t_x1, float* t_ff) {
  using namespace cfg;
  dim3 blk(256);
  const int Mq = B * nq, Mk = B * nkv;
  gemm_bias_kernel<false><<<dim3(DM / 128, ceil_div(Mq, 128)), blk, 0, stream>>>(qin, S.wq, S.bq, t_q, Mq, DM, DM);
  gemm_bias_kernel<false><<<dim3(DM / 128, ceil_div(Mk, 128)), blk, 0, stream>>>(kv, S.wk, S.bk, t_k, Mk, DM, DM);
  gemm_bias_kernel<false><<<dim3(DM / 128, ceil_div(Mk, 128)), blk, 0, stream>>>(kv, S.wv, S.bv, t_v, Mk, DM, DM);
  attn_kernel<<<ceil_div(B * H * nq, 8), blk, 0, stream>>>(t_q, t_k, t_v, geo, mask, t_att,
                                                           nq, nkv, gq0, gk0, mqs, mbs);
  gemm_bias_kernel<false><<<dim3(DM / 128, ceil_div(Mq, 128)), blk, 0, stream>>>(t_att, S.wo, S.bo, t_o, Mq, DM, DM);
  add_ln_kernel<<<ceil_div(Mq, 8), blk, 0, stream>>>(t_o, qin, S.ln1g, S.ln1b, t_x1, Mq);
  gemm_bias_kernel<true><<<dim3(DFF / 128, ceil_div(Mq, 128)), blk, 0, stream>>>(t_x1, S.w1, S.b1, t_ff, Mq, DFF, DM);
  gemm_bias_kernel<false><<<dim3(DM / 128, ceil_div(Mq, 128)), blk, 0, stream>>>(t_ff, S.w2, S.b2, t_o, Mq, DM, DFF);
  add_ln_kernel<<<ceil_div(Mq, 8), blk, 0, stream>>>(t_o, t_x1, S.ln2g, S.ln2b, out, Mq);
}

extern "C" void kernel_launch(void* const* d_in, const int* in_sizes, int n_in,
                              void* d_out, int out_size, void* d_ws, size_t ws_size,
                              hipStream_t stream) {
  using namespace cfg;
  (void)in_sizes; (void)n_in; (void)out_size; (void)ws_size;
  const float* reg_feat  = (const float*)d_in[0];
  const float* reg_box   = (const float*)d_in[1];
  const float* grid_feat = (const float*)d_in[2];
  const float* grid_box  = (const float*)d_in[3];
  // params pytree, dict keys sorted:
  const float* fc_grid_b   = (const float*)d_in[4];
  const float* fc_grid_w   = (const float*)d_in[5];
  const float* fc_gs_b     = (const float*)d_in[6];
  const float* fc_gs_w     = (const float*)d_in[7];
  const float* fc_region_b = (const float*)d_in[8];
  const float* fc_region_w = (const float*)d_in[9];
  const int IN_G2R = 10, IN_GRID = 26, IN_R2G = 46, IN_REGION = 62;
  const float* ln_grid_b   = (const float*)d_in[42];
  const float* ln_grid_g   = (const float*)d_in[43];
  const float* ln_region_b = (const float*)d_in[44];
  const float* ln_region_g = (const float*)d_in[45];

  // workspace bump allocator (256B aligned blocks)
  uintptr_t wp = (uintptr_t)d_ws;
  auto alloc = [&](size_t bytes) -> void* {
    void* p = (void*)wp;
    wp += (bytes + 255) & ~(size_t)255;
    return p;
  };
  unsigned char* pad_r   = (unsigned char*)alloc((size_t)B * NR);
  unsigned char* pad_g   = (unsigned char*)alloc((size_t)B * NG);
  unsigned char* m_r2all = (unsigned char*)alloc((size_t)B * NR * NA);
  unsigned char* m_g2all = (unsigned char*)alloc((size_t)B * NG * NA);
  float* geo   = (float*)alloc((size_t)B * H * NA * NA * 4);
  float* pos_r = (float*)alloc((size_t)NR * DM * 4);
  float* pos_g = (float*)alloc((size_t)NG * DM * 4);
  float* pos_a = (float*)alloc((size_t)NA * DM * 4);
  float* xr    = (float*)alloc((size_t)B * NR * DM * 4);
  float* xg    = (float*)alloc((size_t)B * NG * DM * 4);
  float* xall  = (float*)alloc((size_t)B * NA * DM * 4);
  float* qin_r = (float*)alloc((size_t)B * NR * DM * 4);
  float* qin_g = (float*)alloc((size_t)B * NG * DM * 4);
  float* t_q   = (float*)alloc((size_t)B * NR * DM * 4);
  float* t_k   = (float*)alloc((size_t)B * NA * DM * 4);
  float* t_v   = (float*)alloc((size_t)B * NA * DM * 4);
  float* t_att = (float*)alloc((size_t)B * NR * DM * 4);
  float* t_o   = (float*)alloc((size_t)B * NR * DM * 4);
  float* t_x1  = (float*)alloc((size_t)B * NR * DM * 4);
  float* t_ff  = (float*)alloc((size_t)B * NR * DFF * 4);

  dim3 blk(256);
  // precompute: pads, masks, geometry bias, position embeddings
  pad_kernel<<<ceil_div(B * NR, 8), blk, 0, stream>>>(reg_feat, pad_r, B * NR);
  pad_kernel<<<ceil_div(B * NG, 8), blk, 0, stream>>>(grid_feat, pad_g, B * NG);
  mask_r2all_kernel<<<ceil_div(B * NR * NA, 256), blk, 0, stream>>>(reg_box, pad_r, m_r2all);
  mask_g2all_kernel<<<ceil_div(B * NG * NA, 256), blk, 0, stream>>>(reg_box, pad_g, m_g2all);
  geo_kernel<<<ceil_div(B * NA * NA, 256), blk, 0, stream>>>(reg_box, grid_box, fc_gs_w, fc_gs_b, geo);
  pos_embed_kernel<<<ceil_div(NR * DM, 256), blk, 0, stream>>>(pos_r, NR);
  pos_embed_kernel<<<ceil_div(NG * DM, 256), blk, 0, stream>>>(pos_g, NG);
  pos_embed_kernel<<<ceil_div(NA * DM, 256), blk, 0, stream>>>(pos_a, NA);

  // input projections: LN(ReLU(feat @ W + b))
  gemm_bias_kernel<true><<<dim3(DM / 128, ceil_div(B * NR, 128)), blk, 0, stream>>>(
      reg_feat, fc_region_w, fc_region_b, t_o, B * NR, DM, DIN);
  add_ln_kernel<<<ceil_div(B * NR, 8), blk, 0, stream>>>(t_o, nullptr, ln_region_g, ln_region_b, xr, B * NR);
  gemm_bias_kernel<true><<<dim3(DM / 128, ceil_div(B * NG, 128)), blk, 0, stream>>>(
      grid_feat, fc_grid_w, fc_grid_b, t_o, B * NG, DM, DIN);
  add_ln_kernel<<<ceil_div(B * NG, 8), blk, 0, stream>>>(t_o, nullptr, ln_grid_g, ln_grid_b, xg, B * NG);

  for (int L = 0; L < NL; ++L) {
    StackP pr  = get_stack(d_in, IN_REGION, L);
    StackP pg  = get_stack(d_in, IN_GRID, L);
    StackP prg = get_stack(d_in, IN_R2G, L);
    StackP pgr = get_stack(d_in, IN_G2R, L);

    // region self-attention
    add_pos_kernel<<<ceil_div(B * NR * DM, 256), blk, 0, stream>>>(xr, pos_r, qin_r, B * NR, NR);
    enc_layer(stream, qin_r, qin_r, NR, NR, geo, 0, 0, pad_r, 0, NR, pr, xr,
              t_q, t_k, t_v, t_att, t_o, t_x1, t_ff);
    // grid self-attention
    add_pos_kernel<<<ceil_div(B * NG * DM, 256), blk, 0, stream>>>(xg, pos_g, qin_g, B * NG, NG);
    enc_layer(stream, qin_g, qin_g, NG, NG, geo, NR, NR, pad_g, 0, NG, pg, xg,
              t_q, t_k, t_v, t_att, t_o, t_x1, t_ff);
    // combined KV = concat(region, grid) + pos_all
    concat_pos_kernel<<<ceil_div(B * NA * DM, 256), blk, 0, stream>>>(xr, xg, pos_a, xall);
    // region -> all
    add_pos_kernel<<<ceil_div(B * NR * DM, 256), blk, 0, stream>>>(xr, pos_r, qin_r, B * NR, NR);
    enc_layer(stream, qin_r, xall, NR, NA, geo, 0, 0, m_r2all, NA, NR * NA, prg, xr,
              t_q, t_k, t_v, t_att, t_o, t_x1, t_ff);
    // grid -> all
    add_pos_kernel<<<ceil_div(B * NG * DM, 256), blk, 0, stream>>>(xg, pos_g, qin_g, B * NG, NG);
    enc_layer(stream, qin_g, xall, NG, NA, geo, NR, 0, m_g2all, NA, NG * NA, pgr, xg,
              t_q, t_k, t_v, t_att, t_o, t_x1, t_ff);
  }

  const int total_out = B * NA * DM + B * NA + B * NR * DM;
  finalize_kernel<<<ceil_div(total_out, 256), blk, 0, stream>>>(
      xr, xg, pad_r, pad_g, pos_r, (float*)d_out);
}
